// Correlation3D_78932908966244
// MI455X (gfx1250) — compile-verified, matrix operand order checked by reading, not run on hardware
//
#include <hip/hip_runtime.h>
#include <hip/hip_fp16.h>
#include <stdint.h>

typedef __attribute__((ext_vector_type(2)))  float    v2f;
typedef __attribute__((ext_vector_type(8)))  float    v8f;
typedef __attribute__((ext_vector_type(16))) _Float16 v16h;

// ---------------------------------------------------------------------------
// D = A(16x4) * B(4x16) + C accumulate, K=4 slice.
// Per-lane operands already resolved for this lane's half:
//   lanes 0-15  supply (K0,K1), lanes 16-31 supply (K2,K3)   [ISA 7.12.2]
//   a0/a1 = A[lane&15][kb], A[lane&15][kb+1];  b0/b1 = B[kb][lane&15], B[kb+1][..]
//   with kb = (lane>>4)*2 folded into the *address* by the caller.
// C/D: vgpr r -> M=r (lanes0-15) / M=r+8 (lanes16-31), N=lane&15.
// ---------------------------------------------------------------------------
#define HAS_WMMA_F32_4 __has_builtin(__builtin_amdgcn_wmma_f32_16x16x4_f32)

static __device__ __forceinline__ v8f wmma4(int lane, float a0, float a1,
                                            float b0, float b1, v8f c) {
#if HAS_WMMA_F32_4
  v2f a, b;
  a[0] = a0; a[1] = a1;
  b[0] = b0; b[1] = b1;
  return __builtin_amdgcn_wmma_f32_16x16x4_f32(false, a, false, b, (short)0, c, false, false);
#else
  // Fallback: pad K=4 into 16x16x32 f16; collect K2/K3 from the upper half.
  float a2 = __shfl_xor(a0, 16, 32), a3 = __shfl_xor(a1, 16, 32);
  float b2 = __shfl_xor(b0, 16, 32), b3 = __shfl_xor(b1, 16, 32);
  v16h ah = {};
  v16h bh = {};
  if (lane < 16) {
    ah[0] = (_Float16)a0; ah[1] = (_Float16)a1; ah[2] = (_Float16)a2; ah[3] = (_Float16)a3;
    bh[0] = (_Float16)b0; bh[1] = (_Float16)b1; bh[2] = (_Float16)b2; bh[3] = (_Float16)b3;
  }
  return __builtin_amdgcn_wmma_f32_16x16x32_f16(false, ah, false, bh, (short)0, c, false, false);
#endif
}

// ---------------------------------------------------------------------------
// [B,C,N] -> [B,N,C] LDS-tiled transpose (coalesced both sides).
// ---------------------------------------------------------------------------
__global__ void transpose_kernel(const float* __restrict__ src, float* __restrict__ dst,
                                 int C, int N) {
  __shared__ float tile[32][33];
  const int b = blockIdx.z;
  const int n0 = blockIdx.x * 32, c0 = blockIdx.y * 32;
  const int tx = threadIdx.x, ty = threadIdx.y;  // 32 x 8
  const float* s = src + (size_t)b * C * N;
  float* d = dst + (size_t)b * N * C;
#pragma unroll
  for (int i = 0; i < 32; i += 8)
    tile[ty + i][tx] = s[(size_t)(c0 + ty + i) * N + n0 + tx];
  __syncthreads();
#pragma unroll
  for (int i = 0; i < 32; i += 8)
    d[(size_t)(n0 + ty + i) * C + c0 + tx] = tile[tx][ty + i];
}

// ---------------------------------------------------------------------------
// Brute-force KNN: one thread per query; candidates tiled in LDS as float4 so
// the (wave-uniform address) inner read is a single ds_load_b128 broadcast.
// Register top-K via unrolled bubble insert (strict '<' keeps lowest index on
// ties, matching jax.lax.top_k).  xyz layout: [B,3,N].
// ---------------------------------------------------------------------------
template <int KK>
__global__ void knn_kernel(const float* __restrict__ in_xyz, const float* __restrict__ q_xyz,
                           int Ni, int Nq, int* __restrict__ idx_out) {
  const int b = blockIdx.y;
  const int q = blockIdx.x * blockDim.x + threadIdx.x;
  const float* ib = in_xyz + (size_t)b * 3 * Ni;
  const float* qb = q_xyz + (size_t)b * 3 * Nq;

  const bool valid = (q < Nq);
  float qx = 0.f, qy = 0.f, qz = 0.f;
  if (valid) { qx = qb[q]; qy = qb[Nq + q]; qz = qb[2 * Nq + q]; }

  float bd[KK];
  int   bi[KK];
#pragma unroll
  for (int t = 0; t < KK; ++t) { bd[t] = 3.4e38f; bi[t] = 0; }

  __shared__ float4 sp[256];

  for (int t0 = 0; t0 < Ni; t0 += 256) {
    const int cnt = min(256, Ni - t0);
    for (int i = threadIdx.x; i < cnt; i += blockDim.x)
      sp[i] = make_float4(ib[t0 + i], ib[Ni + t0 + i], ib[2 * Ni + t0 + i], 0.f);
    __syncthreads();
    if (valid) {
      for (int i = 0; i < cnt; ++i) {
        const float4 p = sp[i];
        const float dx = qx - p.x, dy = qy - p.y, dz = qz - p.z;
        const float d = dx * dx + dy * dy + dz * dz;
        if (d < bd[KK - 1]) {
          bd[KK - 1] = d; bi[KK - 1] = t0 + i;
#pragma unroll
          for (int j = KK - 1; j > 0; --j) {
            if (bd[j] < bd[j - 1]) {
              float td = bd[j]; bd[j] = bd[j - 1]; bd[j - 1] = td;
              int   ti = bi[j]; bi[j] = bi[j - 1]; bi[j - 1] = ti;
            }
          }
        }
      }
    }
    __syncthreads();
  }
  if (valid) {
    int* o = idx_out + ((size_t)b * Nq + q) * KK;
#pragma unroll
    for (int t = 0; t < KK; ++t) o[t] = bi[t];
  }
}

// ---------------------------------------------------------------------------
// Transposed-layout pyramid pooling:
//   dst_t[b,m,c] = mean_k src_t[b, idx[b,m,k], c]      (fully coalesced in c)
// ---------------------------------------------------------------------------
__global__ void pool3t_kernel(const float* __restrict__ src_t, const int* __restrict__ idx,
                              float* __restrict__ dst_t, int Nsrc, int Ndst, int Cc, int B) {
  const size_t t = (size_t)blockIdx.x * blockDim.x + threadIdx.x;
  const size_t total = (size_t)B * Ndst * Cc;
  if (t >= total) return;
  const int c = (int)(t % Cc);
  const size_t bm = t / Cc;
  const int m = (int)(bm % Ndst);
  const int b = (int)(bm / Ndst);
  const int* id = idx + ((size_t)b * Ndst + m) * 3;
  const float* s = src_t + (size_t)b * Nsrc * Cc;
  dst_t[t] = (s[(size_t)id[0] * Cc + c] + s[(size_t)id[1] * Cc + c] +
              s[(size_t)id[2] * Cc + c]) * (1.0f / 3.0f);
}

// ---------------------------------------------------------------------------
// Per-level matching cost. One wave per query point:
//  - 16 neighbor correlations: contiguous float4 dots on transposed features
//    (2 lanes per neighbor, 64 ch each, combine via shfl_xor(16))
//  - WMMA w1(16x4) x feat(4x16) + b1, relu
//  - h1 through LDS -> w2(16x16) x h1 + b2 as 4 K=4 WMMAs (2 accumulators)
//  - relu, sum over 16 neighbor columns -> costs[b, 16*lvl+o, n]
// ---------------------------------------------------------------------------
__global__ void level_cost_kernel(const float* __restrict__ xyz1, const float* __restrict__ xyz2,
                                  const float* __restrict__ feat1t, const float* __restrict__ feat2lt,
                                  const int* __restrict__ idxc,
                                  const float* __restrict__ w1, const float* __restrict__ b1,
                                  const float* __restrict__ w2, const float* __restrict__ b2,
                                  float* __restrict__ costs, int Nl, int level) {
  const int N1 = 8192, CIN = 128;
  const int lane = threadIdx.x & 31;
  const int wid = threadIdx.x >> 5;
  const int gw = blockIdx.x * (blockDim.x >> 5) + wid;  // point over B*N1
  const int b = gw / N1;
  const int n = gw % N1;

  __shared__ float hbuf[8][16][17];

  const int r16 = lane & 15;
  const int rb = (lane >> 4) << 3;   // C/D row offset for this half
  const int kb = (lane >> 4) << 1;   // A/B K offset for this half

  // --- gather neighbor + correlation dot (contiguous, vectorized)
  const int nb = idxc[((size_t)b * N1 + n) * 16 + r16];
  const int c0 = (lane >> 4) * 64;
  const float4* f1v = (const float4*)(feat1t + ((size_t)b * N1 + n) * CIN + c0);
  const float4* f2v = (const float4*)(feat2lt + ((size_t)b * Nl + nb) * CIN + c0);
  float acc = 0.f;
#pragma unroll
  for (int c = 0; c < 16; ++c) {
    const float4 u = f1v[c], v = f2v[c];
    acc += u.x * v.x + u.y * v.y + u.z * v.z + u.w * v.w;
  }
  acc += __shfl_xor(acc, 16, 32);
  const float corr = acc * (1.0f / (float)CIN);

  const float qx = xyz1[(size_t)b * 3 * N1 + n];
  const float qy = xyz1[(size_t)b * 3 * N1 + N1 + n];
  const float qz = xyz1[(size_t)b * 3 * N1 + 2 * N1 + n];
  const float px = xyz2[(size_t)b * 3 * Nl + nb] - qx;
  const float py = xyz2[(size_t)b * 3 * Nl + Nl + nb] - qy;
  const float pz = xyz2[(size_t)b * 3 * Nl + 2 * Nl + nb] - qz;

  // --- layer 1: D1 = relu(w1 * feat + b1)
  const float a0 = w1[r16 * 4 + kb];
  const float a1 = w1[r16 * 4 + kb + 1];
  const float b0 = (lane < 16) ? px : pz;
  const float b1v = (lane < 16) ? py : corr;
  v8f c1;
#pragma unroll
  for (int r = 0; r < 8; ++r) c1[r] = b1[r + rb];
  v8f d1 = wmma4(lane, a0, a1, b0, b1v, c1);

  // --- relu + reshape h1 (C/D layout) -> B layout via LDS
#pragma unroll
  for (int r = 0; r < 8; ++r) hbuf[wid][r + rb][r16] = fmaxf(d1[r], 0.f);
  __syncthreads();

  // --- layer 2: K=16 as 4 K=4 WMMAs on two independent accumulators
  v8f cA, cB = {};
#pragma unroll
  for (int r = 0; r < 8; ++r) cA[r] = b2[r + rb];
#pragma unroll
  for (int s = 0; s < 4; ++s) {
    const float wa0 = w2[r16 * 16 + 4 * s + kb];
    const float wa1 = w2[r16 * 16 + 4 * s + kb + 1];
    const float hb0 = hbuf[wid][4 * s + kb][r16];
    const float hb1 = hbuf[wid][4 * s + kb + 1][r16];
    if (s & 1) cB = wmma4(lane, wa0, wa1, hb0, hb1, cB);
    else       cA = wmma4(lane, wa0, wa1, hb0, hb1, cA);
  }
  __syncthreads();

  // --- relu + sum over 16 neighbor columns (xor tree within 16-lane groups)
  float sum[8];
#pragma unroll
  for (int r = 0; r < 8; ++r) {
    float v = fmaxf(cA[r] + cB[r], 0.f);
    v += __shfl_xor(v, 1, 32);
    v += __shfl_xor(v, 2, 32);
    v += __shfl_xor(v, 4, 32);
    v += __shfl_xor(v, 8, 32);
    sum[r] = v;
  }
  if (r16 == 0) {
#pragma unroll
    for (int r = 0; r < 8; ++r)
      costs[((size_t)b * 64 + level * 16 + rb + r) * N1 + n] = sum[r];
  }
}

// ---------------------------------------------------------------------------
// out[b,o,n] = relu(sum_c wm[o,c]*costs[b,c,n] + bm[o])
// One wave per 16x16 tile; K=64 as 16 K=4 WMMAs on two accumulators.
// ---------------------------------------------------------------------------
__global__ void final_matmul_kernel(const float* __restrict__ costs, const float* __restrict__ wm,
                                    const float* __restrict__ bm, float* __restrict__ out) {
  const int N1 = 8192, C = 64;
  const int lane = threadIdx.x & 31;
  const int wid = threadIdx.x >> 5;
  const int gw = blockIdx.x * (blockDim.x >> 5) + wid;  // tile over B*4*(N1/16)
  const int nt = gw % (N1 / 16);
  const int mt = (gw / (N1 / 16)) % 4;
  const int b = gw / ((N1 / 16) * 4);

  const int r16 = lane & 15;
  const int rb = (lane >> 4) << 3;
  const int kb = (lane >> 4) << 1;
  const int row = mt * 16 + r16;
  const int col = nt * 16 + r16;

  v8f cA, cB = {};
#pragma unroll
  for (int r = 0; r < 8; ++r) cA[r] = bm[mt * 16 + rb + r];

  const float* cb = costs + (size_t)b * C * N1;
  const float* wrow = wm + (size_t)row * C;
#pragma unroll
  for (int s = 0; s < 16; ++s) {
    const float a0 = wrow[4 * s + kb];
    const float a1 = wrow[4 * s + kb + 1];
    const float b0 = cb[(size_t)(4 * s + kb) * N1 + col];
    const float b1 = cb[(size_t)(4 * s + kb + 1) * N1 + col];
    if (s & 1) cB = wmma4(lane, a0, a1, b0, b1, cB);
    else       cA = wmma4(lane, a0, a1, b0, b1, cA);
  }
#pragma unroll
  for (int r = 0; r < 8; ++r)
    out[((size_t)b * C + mt * 16 + rb + r) * N1 + col] = fmaxf(cA[r] + cB[r], 0.f);
}

// ---------------------------------------------------------------------------
extern "C" void kernel_launch(void* const* d_in, const int* in_sizes, int n_in,
                              void* d_out, int out_size, void* d_ws, size_t ws_size,
                              hipStream_t stream) {
  (void)in_sizes; (void)n_in; (void)out_size; (void)ws_size;

  const float* xyz1  = (const float*)d_in[0];
  const float* x2[4] = {(const float*)d_in[1], (const float*)d_in[2],
                        (const float*)d_in[3], (const float*)d_in[4]};
  const float* feat1 = (const float*)d_in[5];
  const float* feat2 = (const float*)d_in[6];
  const float* w1 = (const float*)d_in[7];
  const float* b1 = (const float*)d_in[8];
  const float* w2 = (const float*)d_in[9];
  const float* b2 = (const float*)d_in[10];
  const float* wm = (const float*)d_in[11];
  const float* bm = (const float*)d_in[12];
  float* out = (float*)d_out;

  const int B = 2, N1 = 8192, CIN = 128;
  const int Ns[4] = {8192, 2048, 512, 128};

  uint8_t* wsp = (uint8_t*)d_ws;
  auto alloc = [&](size_t bytes) {
    uint8_t* p = wsp;
    wsp += (bytes + 511) & ~(size_t)511;
    return (void*)p;
  };
  float* f1t    = (float*)alloc((size_t)B * N1 * CIN * sizeof(float));     // feat1^T
  float* f2t[4];
  for (int l = 0; l < 4; ++l)
    f2t[l] = (float*)alloc((size_t)B * Ns[l] * CIN * sizeof(float));       // feat2_l^T
  int* idx_p[3];
  for (int l = 0; l < 3; ++l)
    idx_p[l] = (int*)alloc((size_t)B * Ns[l + 1] * 3 * sizeof(int));
  int* idx_c[4];
  for (int l = 0; l < 4; ++l)
    idx_c[l] = (int*)alloc((size_t)B * N1 * 16 * sizeof(int));
  float* costs = (float*)alloc((size_t)B * 64 * N1 * sizeof(float));

  // Point-major feature layouts (coalesced correlation dots).
  transpose_kernel<<<dim3(N1 / 32, CIN / 32, B), dim3(32, 8), 0, stream>>>(feat1, f1t, CIN, N1);
  transpose_kernel<<<dim3(Ns[0] / 32, CIN / 32, B), dim3(32, 8), 0, stream>>>(feat2, f2t[0], CIN, Ns[0]);

  // Pyramid KNN (k=3): query level i against inputs of level i-1.
  knn_kernel<3><<<dim3(Ns[1] / 256, B), 256, 0, stream>>>(x2[0], x2[1], Ns[0], Ns[1], idx_p[0]);
  knn_kernel<3><<<dim3(Ns[2] / 256, B), 256, 0, stream>>>(x2[1], x2[2], Ns[1], Ns[2], idx_p[1]);
  knn_kernel<3><<<dim3(1, B), 128, 0, stream>>>(x2[2], x2[3], Ns[2], Ns[3], idx_p[2]);

  // Pooled (transposed) features: pyramid avg of cost volume == dot with pooled feat2.
  for (int l = 1; l < 4; ++l) {
    const size_t tot = (size_t)B * Ns[l] * CIN;
    pool3t_kernel<<<(unsigned)((tot + 255) / 256), 256, 0, stream>>>(
        f2t[l - 1], idx_p[l - 1], f2t[l], Ns[l - 1], Ns[l], CIN, B);
  }

  // Cross KNN (k=16): query xyz1 against each pyramid level.
  for (int l = 0; l < 4; ++l)
    knn_kernel<16><<<dim3(N1 / 256, B), 256, 0, stream>>>(x2[l], xyz1, Ns[l], N1, idx_c[l]);

  // Per-level WMMA MLP cost.
  for (int l = 0; l < 4; ++l)
    level_cost_kernel<<<(B * N1) / 8, 256, 0, stream>>>(xyz1, x2[l], f1t, f2t[l], idx_c[l],
                                                        w1, b1, w2, b2, costs, Ns[l], l);

  // Final 64x64x8192 WMMA GEMM + bias + relu.
  final_matmul_kernel<<<(B * 4 * (N1 / 16)) / 8, 256, 0, stream>>>(costs, wm, bm, out);
}
